// Attention_42984032699151
// MI455X (gfx1250) — matrix operand order repaired and checked by reading, not compiled, not run on hardware
//
#include <hip/hip_runtime.h>
#include <hip/hip_bf16.h>

// ---------- types ----------
typedef __attribute__((ext_vector_type(16))) __bf16 v16bf;
typedef __attribute__((ext_vector_type(8)))  float  v8f;
typedef __attribute__((ext_vector_type(4)))  unsigned int u32x4;
typedef __attribute__((ext_vector_type(8)))  int i32x8;
typedef __attribute__((ext_vector_type(4)))  int i32x4;

struct __align__(16) U4 { unsigned int x, y, z, w; };

union FragBF {                 // one WMMA 16-bit A/B operand: 8 VGPRs
    v16bf v;
    U4    q[2];
    unsigned short h[16];
};

__device__ __forceinline__ unsigned short f2bf(float f) {
    unsigned int u = __float_as_uint(f);
    unsigned int r = u + 0x7fffu + ((u >> 16) & 1u);   // round-to-nearest-even
    return (unsigned short)(r >> 16);
}

// ---------- Tensor Data Mover (CDNA5) ----------
#if defined(__AMDGCN__) && __has_builtin(__builtin_amdgcn_tensor_load_to_lds) && \
    __has_builtin(__builtin_amdgcn_s_wait_tensorcnt)
#define USE_TDM 1
#else
#define USE_TDM 0
#endif

__device__ __forceinline__ unsigned lds_byte_off(const void* p) {
    // generic LDS address truncates to the DS byte offset (ISA 10.2 aperture rules)
    return (unsigned)(unsigned long long)p;
}

#if USE_TDM
// 2D tile load: tile_x contiguous elements (bf16) per row, tile_y rows,
// row stride stride_x elements. D# packing per ISA 08_async_tensor §8.3/8.4.
__device__ __forceinline__ void tdm_load_2d(unsigned lds_off,
                                            const unsigned short* g,
                                            unsigned tile_x, unsigned tile_y,
                                            unsigned tensor_x, unsigned tensor_y,
                                            unsigned long long stride_x) {
    unsigned long long ga = (unsigned long long)(const void*)g;
    u32x4 g0;
    g0[0] = 1u;                                    // count=1, user descriptor
    g0[1] = lds_off;                               // lds_addr (bytes)
    g0[2] = (unsigned)ga;                          // global_addr[31:0]
    g0[3] = (unsigned)((ga >> 32) & 0x01FFFFFFull) // global_addr[56:32]
          | (2u << 30);                            // type = 2 ("image")
    i32x8 g1;
    g1[0] = (int)(1u << 16);                       // data_size=1 -> 2 bytes
    g1[1] = (int)((tensor_x & 0xFFFFu) << 16);                       // tensor_dim0[15:0]
    g1[2] = (int)((tensor_x >> 16) | ((tensor_y & 0xFFFFu) << 16));  // td0[31:16], td1[15:0]
    g1[3] = (int)((tensor_y >> 16) | (tile_x << 16));                // td1[31:16], tile_dim0
    g1[4] = (int)(tile_y & 0xFFFFu);                                 // tile_dim1 (tile_dim2=0)
    g1[5] = (int)(stride_x & 0xFFFFFFFFull);                         // tensor_dim0_stride lo
    g1[6] = (int)((stride_x >> 32) & 0xFFFFull);                     // stride hi16
    g1[7] = 0;
    i32x4 z4 = {};
#if __clang_major__ >= 23
    i32x8 z8 = {};
    __builtin_amdgcn_tensor_load_to_lds(g0, g1, z4, z4, z8, 0);
#else
    __builtin_amdgcn_tensor_load_to_lds(g0, g1, z4, z4, 0);
#endif
}
#endif

// ---------- generic f32 -> bf16 cast ----------
__global__ void cast_f32_bf16(const float* __restrict__ src,
                              unsigned short* __restrict__ dst, long n) {
    long i = blockIdx.x * (long)blockDim.x + threadIdx.x;
    long stride = (long)gridDim.x * blockDim.x;
    for (; i < n; i += stride) dst[i] = f2bf(src[i]);
}

// ---------- transpose + cast: W[K][N] f32 -> WT[N][K] bf16 ----------
__global__ void transpose_cast_w(const float* __restrict__ src,
                                 unsigned short* __restrict__ dst,
                                 int K, int N) {
    long total = (long)K * N;
    long i = blockIdx.x * (long)blockDim.x + threadIdx.x;
    long stride = (long)gridDim.x * blockDim.x;
    for (; i < total; i += stride) {
        int  n = (int)(i % N);
        long k = i / N;
        dst[(size_t)n * K + k] = f2bf(src[i]);
    }
}

// ---------- transpose + cast V: [2048][1024] f32 -> [1024][2048] bf16 ------
__global__ void transpose_cast_v(const float* __restrict__ src,
                                 unsigned short* __restrict__ dst) {
    long total = 2048l * 1024;
    long i = blockIdx.x * (long)blockDim.x + threadIdx.x;
    long stride = (long)gridDim.x * blockDim.x;
    for (; i < total; i += stride) {
        int c = (int)(i & 1023);     // channel (kvh*128 + d)
        int t = (int)(i >> 10);      // time
        dst[(size_t)c * 2048 + t] = f2bf(src[i]);
    }
}

// ---------- RoPE + cast (q or k) ----------
__global__ void rope_cast(const float* __restrict__ src,
                          const float* __restrict__ cosb,
                          const float* __restrict__ sinb,
                          unsigned short* __restrict__ dst, int nheads) {
    long idx = blockIdx.x * (long)blockDim.x + threadIdx.x;
    long total = (long)2048 * nheads * 64;
    if (idx >= total) return;
    int  i = (int)(idx & 63);
    int  h = (int)((idx >> 6) % nheads);
    long t = idx / (64 * (long)nheads);
    size_t base = (size_t)t * nheads * 128 + (size_t)h * 128 + 2 * i;
    float a = src[base], b = src[base + 1];
    float c = cosb[t * 64 + i], s = sinb[t * 64 + i];
    dst[base]     = f2bf(a * c - b * s);
    dst[base + 1] = f2bf(a * s + b * c);
}

// ---------- tiled WMMA GEMM: C[M,N] = A[M,K] * BT[N,K]^T, bf16 in, f32 out --
// 256 threads = 8 waves; workgroup tile 128x128; wave tile 32x64; k-step 32.
// Both LDS tiles are plain row-major 2D tiles -> staged by TDM, double-buffered.
__global__ __launch_bounds__(256)
void gemm_bf16(const unsigned short* __restrict__ A,   // [M][K]
               const unsigned short* __restrict__ BT,  // [N][K]
               float* __restrict__ C, int M, int N, int K) {
    __shared__ __align__(16) unsigned short As[2][128 * 32];  // [m][k]
    __shared__ __align__(16) unsigned short Bs[2][128 * 32];  // [n][k]

    const int tid  = threadIdx.x;
    const int m0   = blockIdx.x * 128;
    const int n0   = blockIdx.y * 128;
    const int wid  = tid >> 5, lane = tid & 31;
    const int hf   = lane >> 4, ln = lane & 15;
    const int wr   = wid >> 1,  wc = wid & 1;       // 4x2 wave grid

    const v8f vzero = {};
    v8f acc[2][4];
    #pragma unroll
    for (int mt = 0; mt < 2; ++mt)
        #pragma unroll
        for (int nt = 0; nt < 4; ++nt) acc[mt][nt] = vzero;

#if USE_TDM
    if (wid == 0) {   // prologue: fill buffer 0
        tdm_load_2d(lds_byte_off(As[0]), A  + (size_t)m0 * K, 32, 128, K, M, K);
        tdm_load_2d(lds_byte_off(Bs[0]), BT + (size_t)n0 * K, 32, 128, K, N, K);
    }
#endif

    int cur = 0;
    for (int kk = 0; kk < K; kk += 32, cur ^= 1) {
#if USE_TDM
        if (wid == 0) __builtin_amdgcn_s_wait_tensorcnt(0);
        __syncthreads();                       // tile `cur` visible, buf cur^1 free
        if (wid == 0 && kk + 32 < K) {
            tdm_load_2d(lds_byte_off(As[cur ^ 1]),
                        A  + (size_t)m0 * K + kk + 32, 32, 128, K, M, K);
            tdm_load_2d(lds_byte_off(Bs[cur ^ 1]),
                        BT + (size_t)n0 * K + kk + 32, 32, 128, K, N, K);
        }
#else
        __syncthreads();
        {   // fallback: synchronous b128 staging (both tiles row-major)
            int row = tid >> 1, c0 = (tid & 1) * 16;
            const U4* sa = (const U4*)(A  + (size_t)(m0 + row) * K + kk + c0);
            const U4* sb = (const U4*)(BT + (size_t)(n0 + row) * K + kk + c0);
            U4 a0 = sa[0], a1 = sa[1], b0 = sb[0], b1 = sb[1];
            *(U4*)(&As[cur][row * 32 + c0])     = a0;
            *(U4*)(&As[cur][row * 32 + c0 + 8]) = a1;
            *(U4*)(&Bs[cur][row * 32 + c0])     = b0;
            *(U4*)(&Bs[cur][row * 32 + c0 + 8]) = b1;
        }
        if (kk + 32 < K)
            __builtin_prefetch(A + (size_t)(m0 + (tid >> 1)) * K + kk + 32, 0, 1);
        __syncthreads();
#endif
        // A fragments: lane M = ln; VGPR0-3 = K[hf*8..+7], VGPR4-7 = K[16+hf*8..+7]
        FragBF afr[2], bfr[4];
        #pragma unroll
        for (int mt = 0; mt < 2; ++mt) {
            const unsigned short* base = &As[cur][(wr * 32 + mt * 16 + ln) * 32];
            afr[mt].q[0] = *(const U4*)(base + hf * 8);
            afr[mt].q[1] = *(const U4*)(base + 16 + hf * 8);
        }
        // B fragments: lane N = ln; VGPR0-3 = K[hf*8..+7], VGPR4-7 = K[16+hf*8..+7]
        #pragma unroll
        for (int nt = 0; nt < 4; ++nt) {
            const unsigned short* base = &Bs[cur][(wc * 64 + nt * 16 + ln) * 32 + hf * 8];
            bfr[nt].q[0] = *(const U4*)(base);
            bfr[nt].q[1] = *(const U4*)(base + 16);
        }
        #pragma unroll
        for (int mt = 0; mt < 2; ++mt)
            #pragma unroll
            for (int nt = 0; nt < 4; ++nt)
                acc[mt][nt] = __builtin_amdgcn_wmma_f32_16x16x32_bf16(
                    false, afr[mt].v, false, bfr[nt].v,
                    (short)0, acc[mt][nt], false, false);
    }

    // C layout: VGPR r -> row M = r + 8*hf, col N = ln
    #pragma unroll
    for (int mt = 0; mt < 2; ++mt)
        #pragma unroll
        for (int nt = 0; nt < 4; ++nt) {
            int col = n0 + wc * 64 + nt * 16 + ln;
            #pragma unroll
            for (int r = 0; r < 8; ++r) {
                int row = m0 + wr * 32 + mt * 16 + hf * 8 + r;
                C[(size_t)row * N + col] = acc[mt][nt][r];
            }
        }
}

// ---------- flash attention (causal, GQA 32q/8kv, d=128) ----------
// grid (T/64, 32 heads), 128 threads = 4 waves, each wave owns 16 q rows.
__global__ __launch_bounds__(128)
void attn_kernel(const unsigned short* __restrict__ Q,   // [2048][32*128] bf16
                 const unsigned short* __restrict__ Kg,  // [2048][8*128]
                 const unsigned short* __restrict__ VTg, // [8*128][2048] (transposed)
                 unsigned short* __restrict__ Ctx) {     // [2048][32*128]
    __shared__ __align__(16) unsigned short Kl[64 * 128];   // [key][d]
    __shared__ __align__(16) unsigned short Vt[128 * 64];   // [d][key]
    __shared__ __align__(16) unsigned short Pl[4 * 16 * 64];// per-wave P tile

    const int tid  = threadIdx.x;
    const int w    = tid >> 5, lane = tid & 31;
    const int hf   = lane >> 4, ln = lane & 15;
    const int qt   = blockIdx.x;
    const int head = blockIdx.y;
    const int kvh  = head >> 2;
    const int qbase = qt * 64;
    const int wq0   = qbase + w * 16;

    // Q fragments resident: 16 rows x 128 d per wave (4 k-chunks of 32)
    FragBF qf[4];
    {
        const unsigned short* qp = Q + (size_t)(wq0 + ln) * 4096 + head * 128;
        #pragma unroll
        for (int kc = 0; kc < 4; ++kc) {
            qf[kc].q[0] = *(const U4*)(qp + kc * 32 + hf * 8);
            qf[kc].q[1] = *(const U4*)(qp + kc * 32 + 16 + hf * 8);
        }
    }

    const v8f vzero = {};
    float m_r[8], l_r[8];
    v8f o[8];
    #pragma unroll
    for (int r = 0; r < 8; ++r) { m_r[r] = -1e30f; l_r[r] = 0.f; }
    #pragma unroll
    for (int dt = 0; dt < 8; ++dt) o[dt] = vzero;

    const float sc = 0.08838834764831845f;   // 1/sqrt(128)
    const int nkb = qt + 1;                  // causal: only blocks <= q tile

    for (int kb = 0; kb < nkb; ++kb) {
        __syncthreads();                     // previous tiles fully consumed
#if USE_TDM
        if (w == 0) {
            // K block: 64 keys x 128 d, row-major in global
            tdm_load_2d(lds_byte_off(Kl),
                        Kg + (size_t)(kb * 64) * 1024 + kvh * 128,
                        128, 64, 1024, 2048, 1024);
            // V block from pre-transposed global: 128 d rows x 64 keys
            tdm_load_2d(lds_byte_off(Vt),
                        VTg + (size_t)(kvh * 128) * 2048 + kb * 64,
                        64, 128, 2048, 1024, 2048);
            __builtin_amdgcn_s_wait_tensorcnt(0);
        }
#else
        #pragma unroll
        for (int j = 0; j < 8; ++j) {
            int u = tid + j * 128;
            int key = u >> 4, x = u & 15;    // K tile: 64 x (16 segs of 8)
            *(U4*)(&Kl[key * 128 + x * 8]) =
                *(const U4*)(Kg + (size_t)(kb * 64 + key) * 1024 + kvh * 128 + x * 8);
            int d = u >> 3, sg = u & 7;      // V tile: 128 x (8 segs of 8)
            *(U4*)(&Vt[d * 64 + sg * 8]) =
                *(const U4*)(VTg + (size_t)(kvh * 128 + d) * 2048 + kb * 64 + sg * 8);
        }
#endif
        __syncthreads();                     // tiles visible to all waves

        // S = Q * K^T : 16x64 per wave (4 tiles of 16x16)
        v8f s[4];
        #pragma unroll
        for (int kt = 0; kt < 4; ++kt) s[kt] = vzero;
        #pragma unroll
        for (int kt = 0; kt < 4; ++kt) {
            const unsigned short* kbase = Kl + (kt * 16 + ln) * 128;
            #pragma unroll
            for (int kc = 0; kc < 4; ++kc) {
                FragBF bf;
                bf.q[0] = *(const U4*)(kbase + kc * 32 + hf * 8);
                bf.q[1] = *(const U4*)(kbase + kc * 32 + 16 + hf * 8);
                s[kt] = __builtin_amdgcn_wmma_f32_16x16x32_bf16(
                    false, qf[kc].v, false, bf.v, (short)0, s[kt], false, false);
            }
        }

        // scale + causal mask (only the diagonal block needs masking)
        const bool diag = (kb == qt);
        #pragma unroll
        for (int kt = 0; kt < 4; ++kt)
            #pragma unroll
            for (int r = 0; r < 8; ++r) {
                float v = s[kt][r] * sc;
                if (diag) {
                    int kcol = kb * 64 + kt * 16 + ln;
                    int qrow = wq0 + hf * 8 + r;
                    if (kcol > qrow) v = -1e30f;
                }
                s[kt][r] = v;
            }

        // online softmax
        float rm[8];
        #pragma unroll
        for (int r = 0; r < 8; ++r) {
            float v = fmaxf(fmaxf(s[0][r], s[1][r]), fmaxf(s[2][r], s[3][r]));
            #pragma unroll
            for (int off = 1; off < 16; off <<= 1)
                v = fmaxf(v, __shfl_xor(v, off, 32));
            rm[r] = v;
        }

        float alpha[8], rsum[8];
        unsigned short* pl = Pl + w * 1024;
        #pragma unroll
        for (int r = 0; r < 8; ++r) {
            float mn = fmaxf(m_r[r], rm[r]);
            alpha[r] = __expf(m_r[r] - mn);
            m_r[r] = mn;
            float rs = 0.f;
            #pragma unroll
            for (int kt = 0; kt < 4; ++kt) {
                float p = __expf(s[kt][r] - mn);
                rs += p;
                pl[(hf * 8 + r) * 64 + kt * 16 + ln] = f2bf(p);  // same-wave LDS
            }
            rsum[r] = rs;
        }
        #pragma unroll
        for (int r = 0; r < 8; ++r) {
            float v = rsum[r];
            #pragma unroll
            for (int off = 1; off < 16; off <<= 1)
                v += __shfl_xor(v, off, 32);
            l_r[r] = l_r[r] * alpha[r] + v;
        }

        // rescale O, then O += P * V via WMMA
        #pragma unroll
        for (int dt = 0; dt < 8; ++dt)
            #pragma unroll
            for (int r = 0; r < 8; ++r) o[dt][r] *= alpha[r];

        #pragma unroll
        for (int kc = 0; kc < 2; ++kc) {
            FragBF af;
            const unsigned short* pb = pl + ln * 64 + kc * 32;
            af.q[0] = *(const U4*)(pb + hf * 8);
            af.q[1] = *(const U4*)(pb + 16 + hf * 8);
            #pragma unroll
            for (int dt = 0; dt < 8; ++dt) {
                FragBF bf;
                const unsigned short* vb = Vt + (dt * 16 + ln) * 64 + kc * 32 + hf * 8;
                bf.q[0] = *(const U4*)(vb);
                bf.q[1] = *(const U4*)(vb + 16);
                o[dt] = __builtin_amdgcn_wmma_f32_16x16x32_bf16(
                    false, af.v, false, bf.v, (short)0, o[dt], false, false);
            }
        }
    }

    // epilogue: O / l -> ctx (bf16)
    #pragma unroll
    for (int r = 0; r < 8; ++r) {
        float inv = 1.0f / l_r[r];
        int qrow = wq0 + hf * 8 + r;
        #pragma unroll
        for (int dt = 0; dt < 8; ++dt)
            Ctx[(size_t)qrow * 4096 + head * 128 + dt * 16 + ln] =
                f2bf(o[dt][r] * inv);
    }
}

// ---------- launch ----------
extern "C" void kernel_launch(void* const* d_in, const int* in_sizes, int n_in,
                              void* d_out, int out_size, void* d_ws, size_t ws_size,
                              hipStream_t stream) {
    const float* x   = (const float*)d_in[0];
    const float* rc  = (const float*)d_in[1];
    const float* rsn = (const float*)d_in[2];
    const float* wq  = (const float*)d_in[3];
    const float* wk  = (const float*)d_in[4];
    const float* wv  = (const float*)d_in[5];
    const float* wo  = (const float*)d_in[6];
    float* out = (float*)d_out;

    char* ws = (char*)d_ws;
    size_t off = 0;
    auto alloc = [&](size_t bytes) -> void* {
        void* p = ws + off;
        off += (bytes + 255) & ~(size_t)255;
        return p;
    };
    unsigned short* xb   = (unsigned short*)alloc(2048ull * 4096 * 2);
    unsigned short* wqT  = (unsigned short*)alloc(4096ull * 4096 * 2);  // [N][K]
    unsigned short* wkT  = (unsigned short*)alloc(1024ull * 4096 * 2);
    unsigned short* wvT  = (unsigned short*)alloc(1024ull * 4096 * 2);
    unsigned short* woT  = (unsigned short*)alloc(4096ull * 4096 * 2);
    float*          qf   = (float*)alloc(2048ull * 4096 * 4);
    float*          kf   = (float*)alloc(2048ull * 1024 * 4);
    float*          vf   = (float*)alloc(2048ull * 1024 * 4);
    unsigned short* qbb  = (unsigned short*)alloc(2048ull * 4096 * 2);
    unsigned short* kbb  = (unsigned short*)alloc(2048ull * 1024 * 2);
    unsigned short* vtg  = (unsigned short*)alloc(1024ull * 2048 * 2);  // [8*128][2048]
    unsigned short* ctxb = (unsigned short*)alloc(2048ull * 4096 * 2);

    // 1) casts (weights transposed to [N][K] so all GEMM tiles are row-major)
    cast_f32_bf16  <<<2048, 256, 0, stream>>>(x, xb, 2048ll * 4096);
    transpose_cast_w<<<8192, 256, 0, stream>>>(wq, wqT, 4096, 4096);
    transpose_cast_w<<<4096, 256, 0, stream>>>(wk, wkT, 4096, 1024);
    transpose_cast_w<<<4096, 256, 0, stream>>>(wv, wvT, 4096, 1024);
    transpose_cast_w<<<8192, 256, 0, stream>>>(wo, woT, 4096, 4096);

    // 2) projection GEMMs
    gemm_bf16<<<dim3(16, 32), 256, 0, stream>>>(xb, wqT, qf, 2048, 4096, 4096);
    gemm_bf16<<<dim3(16, 8),  256, 0, stream>>>(xb, wkT, kf, 2048, 1024, 4096);
    gemm_bf16<<<dim3(16, 8),  256, 0, stream>>>(xb, wvT, vf, 2048, 1024, 4096);

    // 3) RoPE(q,k) + transpose-cast v
    rope_cast<<<(2048 * 32 * 64) / 256, 256, 0, stream>>>(qf, rc, rsn, qbb, 32);
    rope_cast<<<(2048 * 8 * 64) / 256,  256, 0, stream>>>(kf, rc, rsn, kbb, 8);
    transpose_cast_v<<<4096, 256, 0, stream>>>(vf, vtg);

    // 4) flash attention
    attn_kernel<<<dim3(2048 / 64, 32), 128, 0, stream>>>(qbb, kbb, vtg, ctxb);

    // 5) output projection -> f32 result
    gemm_bf16<<<dim3(16, 32), 256, 0, stream>>>(ctxb, woT, out, 2048, 4096, 4096);
}
